// Attention_82205674045790
// MI455X (gfx1250) — compile-verified, matrix operand order checked by reading, not run on hardware
//
#include <hip/hip_runtime.h>
#include <stdint.h>

// ---- CDNA5 WMMA / TDM types -------------------------------------------------
typedef __attribute__((ext_vector_type(16))) __bf16        v16bf;
typedef __attribute__((ext_vector_type(8)))  float         v8f;
typedef __attribute__((ext_vector_type(4)))  unsigned int  v4u;
typedef __attribute__((ext_vector_type(4)))  unsigned int  u32x4;
typedef __attribute__((ext_vector_type(8)))  int           i32x8;
typedef __attribute__((ext_vector_type(4)))  int           i32x4;

union FragBF {
    v16bf v;
    v4u   u[2];
};
union BF4 {
    __bf16             b[4];
    unsigned long long q;
};
union BF8 {
    v4u    u;
    __bf16 b[8];
};

__device__ __forceinline__ v8f vzero8() {
    v8f z;
    #pragma unroll
    for (int i = 0; i < 8; ++i) z[i] = 0.0f;
    return z;
}

// LDS byte offset of a __shared__ object (addrspacecast generic -> AS3, then ptrtoint)
__device__ __forceinline__ unsigned lds_off(const void* p) {
    return (unsigned)(uintptr_t)(const __attribute__((address_space(3))) void*)p;
}

// ---- Tensor Data Mover: 2D bf16 tile (tileD1 rows x tileD0 elems) -> LDS ----
// D# per CDNA5 ISA. Issued by one wave; tracked by TENSORcnt.
// This toolchain exposes the 6-arg builtin: (g0, g1, g2, g3, g4, cpol).
__device__ __forceinline__ void tdm_load_2d_bf16(const void* gsrc, unsigned ldsDst,
                                                 int tileD0, int tileD1, int stride0) {
    const unsigned long long ga = (unsigned long long)(uintptr_t)gsrc;
    u32x4 g0;
    g0[0] = 1u;                                                  // count=1 (valid), user mode
    g0[1] = ldsDst;                                              // lds_addr (bytes)
    g0[2] = (unsigned)ga;                                        // global_addr[31:0]
    g0[3] = (unsigned)((ga >> 32) & 0x01FFFFFFu) | (2u << 30);   // global_addr[56:32], type=2
    i32x8 g1;
    g1[0] = 1 << 16;                                             // data_size=1 (2 bytes), mask=0
    g1[1] = (tileD0 & 0xFFFF) << 16;                             // tensor_dim0[15:0] @bits63:48
    g1[2] = (tileD1 & 0xFFFF) << 16;                             // tensor_dim1[15:0] @bits95:80
    g1[3] = (tileD0 & 0xFFFF) << 16;                             // tile_dim0 @bits127:112
    g1[4] = (tileD1 & 0xFFFF);                                   // tile_dim1 @bits143:128
    g1[5] = stride0;                                             // tensor_dim0_stride[31:0]
    g1[6] = 0;
    g1[7] = 0;
    i32x4 gz;
    gz[0] = 0; gz[1] = 0; gz[2] = 0; gz[3] = 0;                  // 2D: groups 2/3 unused
    i32x8 gz8;
    #pragma unroll
    for (int i = 0; i < 8; ++i) gz8[i] = 0;
    __builtin_amdgcn_tensor_load_to_lds(g0, g1, gz, gz, gz8, 0);
}

// ---- GEMM: C[M,N] = A[M,K] @ W[N,K]^T, bf16 WMMA ---------------------------
// Block: 256 threads = 8 waves. Block tile 256x128, per-wave 32x128.
// MODE 0: store f32 to outF[M*N]
// MODE 1: RoPE + bf16 store to outB laid out [B,H,L,HD]   (q / k)
// MODE 2: bf16 store to outB laid out [B,H,L,HD]          (v)
// ABF  0: A operand is f32 (convert while staging);  1: A is bf16 (straight copy)
#define BLK_M 256
#define BLK_N 128
#define BLK_K 32

template<int MODE, int ABF>
__global__ void __launch_bounds__(256)
gemm_bt_kernel(const void* __restrict__ Araw, const float* __restrict__ W,
               float* __restrict__ outF, __bf16* __restrict__ outB,
               const float* __restrict__ cosT, const float* __restrict__ sinT,
               int M, int N, int K, float outScale)
{
    __shared__ __align__(16) __bf16 As[BLK_M * BLK_K];   // 16 KB
    __shared__ __align__(16) __bf16 Bs[BLK_N * BLK_K];   //  8 KB

    const int tid      = threadIdx.x;
    const int lane     = tid & 31;
    const int wave     = tid >> 5;
    const int laneIdx  = lane & 15;
    const int khalf    = (lane >> 4) * 8;
    const int rowBlock = blockIdx.y * BLK_M;
    const int colBlock = blockIdx.x * BLK_N;

    const float*  Af = (const float*)Araw;
    const __bf16* Ab = (const __bf16*)Araw;

    v8f acc0[8], acc1[8];   // two 16-row groups: rows wave*32+0..15 and +16..31
    #pragma unroll
    for (int i = 0; i < 8; ++i) { acc0[i] = vzero8(); acc1[i] = vzero8(); }

    const int kTiles = K / BLK_K;
    for (int kt = 0; kt < kTiles; ++kt) {
        const int kBase = kt * BLK_K;

        // ---- stage A tile (256 x 32) ----
        if (ABF == 0) {
            #pragma unroll
            for (int i = tid; i < BLK_M * (BLK_K / 4); i += 256) {   // 2048 float4s
                const int r  = i >> 3;
                const int c4 = (i & 7) * 4;
                const float4 v = *(const float4*)(Af + (size_t)(rowBlock + r) * K + kBase + c4);
                BF4 t;
                t.b[0] = (__bf16)v.x; t.b[1] = (__bf16)v.y;
                t.b[2] = (__bf16)v.z; t.b[3] = (__bf16)v.w;
                *(unsigned long long*)(As + r * BLK_K + c4) = t.q;
            }
        } else {
            #pragma unroll
            for (int i = tid; i < BLK_M * (BLK_K / 8); i += 256) {   // 1024 16B copies
                const int r  = i >> 2;
                const int c8 = (i & 3) * 8;
                *(v4u*)(As + r * BLK_K + c8) =
                    *(const v4u*)(Ab + (size_t)(rowBlock + r) * K + kBase + c8);
            }
        }
        // ---- stage B tile (128 x 32), rows of W are the N dimension ----
        #pragma unroll
        for (int i = tid; i < BLK_N * (BLK_K / 4); i += 256) {       // 1024 float4s
            const int r  = i >> 3;
            const int c4 = (i & 7) * 4;
            const float4 v = *(const float4*)(W + (size_t)(colBlock + r) * K + kBase + c4);
            BF4 t;
            t.b[0] = (__bf16)v.x; t.b[1] = (__bf16)v.y;
            t.b[2] = (__bf16)v.z; t.b[3] = (__bf16)v.w;
            *(unsigned long long*)(Bs + r * BLK_K + c4) = t.q;
        }
        // prefetch next K tile into caches (global_prefetch_b8)
        if (kt + 1 < kTiles) {
            const int c = (tid & 1) * 16;
            if (ABF == 0)
                __builtin_prefetch(Af + (size_t)(rowBlock + (tid >> 1)) * K + kBase + BLK_K + c, 0, 3);
            else
                __builtin_prefetch(Ab + (size_t)(rowBlock + (tid >> 1)) * K + kBase + BLK_K + c, 0, 3);
            if (tid < 128)
                __builtin_prefetch(W + (size_t)(colBlock + tid) * K + kBase + BLK_K, 0, 3);
        }
        __syncthreads();

        // ---- two A fragments: rows wave*32 + (0..15) and + (16..31) ----
        FragBF a0, a1;
        {
            const __bf16* p0 = As + (wave * 32 + laneIdx) * BLK_K + khalf;
            a0.u[0] = *(const v4u*)(p0);
            a0.u[1] = *(const v4u*)(p0 + 16);
            const __bf16* p1 = As + (wave * 32 + 16 + laneIdx) * BLK_K + khalf;
            a1.u[0] = *(const v4u*)(p1);
            a1.u[1] = *(const v4u*)(p1 + 16);
        }
        #pragma unroll
        for (int nt = 0; nt < 8; ++nt) {
            FragBF bf;
            const __bf16* p = Bs + (nt * 16 + laneIdx) * BLK_K + khalf;
            bf.u[0] = *(const v4u*)(p);
            bf.u[1] = *(const v4u*)(p + 16);
            acc0[nt] = __builtin_amdgcn_wmma_f32_16x16x32_bf16(
                false, a0.v, false, bf.v, (short)0, acc0[nt], false, false);
            acc1[nt] = __builtin_amdgcn_wmma_f32_16x16x32_bf16(
                false, a1.v, false, bf.v, (short)0, acc1[nt], false, false);
        }
        __syncthreads();
    }

    // ---- epilogue (two row groups) ----
    #pragma unroll
    for (int g = 0; g < 2; ++g) {
        #pragma unroll
        for (int nt = 0; nt < 8; ++nt) {
            const int gCol = colBlock + nt * 16 + laneIdx;
            #pragma unroll
            for (int r = 0; r < 8; ++r) {
                const int gRow = rowBlock + wave * 32 + g * 16 + r + (lane >> 4) * 8;
                float val = (g ? acc1[nt][r] : acc0[nt][r]) * outScale;
                if (MODE == 0) {
                    outF[(size_t)gRow * N + gCol] = val;
                } else {
                    const int b = gRow >> 11;       // / 2048
                    const int l = gRow & 2047;
                    const int h = gCol >> 7;
                    const int d = gCol & 127;
                    if (MODE == 1) {
                        const float c = cosT[l * 64 + (d >> 1)];
                        const float s = sinT[l * 64 + (d >> 1)];
                        const float partner = __shfl_xor(val, 1, 32);
                        val = (d & 1) ? (partner * s + val * c)
                                      : (val * c - partner * s);
                    }
                    outB[(((size_t)(b * 32 + h) * 2048) + l) * 128 + d] = (__bf16)val;
                }
            }
        }
    }
}

// ---- Flash attention per head: Out[b,l,h*128+d] (bf16) ----------------------
// block = 256 threads (8 waves), 128 query rows, keys in chunks of 32.
// K tiles staged by the Tensor Data Mover (double-buffered, TENSORcnt overlap);
// V tiles staged manually with transpose (TDM cannot transpose), double-buffered.
__global__ void __launch_bounds__(256)
attn_kernel(const __bf16* __restrict__ Q,
            const __bf16* __restrict__ Kb,
            const __bf16* __restrict__ Vb,
            __bf16* __restrict__ Out)
{
    __shared__ __align__(16) __bf16 Ks[2][32 * 128];    // [buf][key][d]
    __shared__ __align__(16) __bf16 Vs[2][128 * 32];    // [buf][d][key] (transposed)
    __shared__ __align__(16) __bf16 Ps[8 * 16 * 32];    // per-wave P tiles

    const int tid     = threadIdx.x;
    const int lane    = tid & 31;
    const int wave    = tid >> 5;
    const int laneIdx = lane & 15;
    const int khalf   = (lane >> 4) * 8;
    const int bh      = blockIdx.x;          // b*32 + h
    const int b       = bh >> 5;
    const int h       = bh & 31;
    const int qBase   = blockIdx.y * 128;

    const size_t headOff = (size_t)bh * 2048 * 128;
    const unsigned ldsK[2] = { lds_off(&Ks[0][0]), lds_off(&Ks[1][0]) };

    // Q fragments for this wave's 16 rows: 16 x 128 = 4 fragments, in registers
    v16bf qfrag[4];
    {
        const int qRow = qBase + wave * 16 + laneIdx;
        const __bf16* qp = Q + headOff + (size_t)qRow * 128;
        #pragma unroll
        for (int kk = 0; kk < 4; ++kk) {
            FragBF f;
            const __bf16* p = qp + kk * 32 + khalf;
            f.u[0] = *(const v4u*)(p);
            f.u[1] = *(const v4u*)(p + 16);
            qfrag[kk] = f.v;
        }
    }

    v8f O[8];
    float mrow[8], lrow[8];
    #pragma unroll
    for (int i = 0; i < 8; ++i) { O[i] = vzero8(); mrow[i] = -1e30f; lrow[i] = 0.0f; }

    __bf16* Pw = Ps + wave * 512;

    // ---- prologue: stage tile 0 ----
    if (wave == 0)
        tdm_load_2d_bf16(Kb + headOff, ldsK[0], 128, 32, 128);
    {
        const __bf16* vp = Vb + headOff;
        for (int i = tid; i < 512; i += 256) {
            const int key = i >> 4;
            const int ch  = (i & 15) * 8;
            BF8 raw;
            raw.u = *(const v4u*)(vp + key * 128 + ch);
            #pragma unroll
            for (int j = 0; j < 8; ++j) Vs[0][(ch + j) * 32 + key] = raw.b[j];
        }
    }
    if (wave == 0) __builtin_amdgcn_s_wait_tensorcnt(0);
    __syncthreads();

    for (int kb = 0; kb < 64; ++kb) {
        const int cur = kb & 1;

        // ---- stage next tile into the other buffer (overlaps compute) ----
        if (kb + 1 < 64) {
            if (wave == 0)
                tdm_load_2d_bf16(Kb + headOff + (size_t)(kb + 1) * 32 * 128,
                                 ldsK[cur ^ 1], 128, 32, 128);
            const __bf16* vp = Vb + headOff + (size_t)(kb + 1) * 32 * 128;
            for (int i = tid; i < 512; i += 256) {
                const int key = i >> 4;
                const int ch  = (i & 15) * 8;
                BF8 raw;
                raw.u = *(const v4u*)(vp + key * 128 + ch);
                #pragma unroll
                for (int j = 0; j < 8; ++j) Vs[cur ^ 1][(ch + j) * 32 + key] = raw.b[j];
            }
        }

        // ---- scores S = Q @ K^T for 16 rows x 32 keys (two 16x16 tiles) ----
        const __bf16* Kc = &Ks[cur][0];
        v8f s0 = vzero8(), s1 = vzero8();
        #pragma unroll
        for (int kk = 0; kk < 4; ++kk) {
            {
                FragBF f;
                const __bf16* p = Kc + (size_t)(laneIdx) * 128 + kk * 32 + khalf;
                f.u[0] = *(const v4u*)(p); f.u[1] = *(const v4u*)(p + 16);
                s0 = __builtin_amdgcn_wmma_f32_16x16x32_bf16(
                    false, qfrag[kk], false, f.v, (short)0, s0, false, false);
            }
            {
                FragBF f;
                const __bf16* p = Kc + (size_t)(16 + laneIdx) * 128 + kk * 32 + khalf;
                f.u[0] = *(const v4u*)(p); f.u[1] = *(const v4u*)(p + 16);
                s1 = __builtin_amdgcn_wmma_f32_16x16x32_bf16(
                    false, qfrag[kk], false, f.v, (short)0, s1, false, false);
            }
        }

        // ---- online softmax per row (rows live across 16-lane halves) ----
        #pragma unroll
        for (int r = 0; r < 8; ++r) {
            float mx = fmaxf(s0[r], s1[r]);
            #pragma unroll
            for (int off = 8; off >= 1; off >>= 1)
                mx = fmaxf(mx, __shfl_xor(mx, off, 32));
            const float mnew  = fmaxf(mrow[r], mx);
            const float scale = __expf(mrow[r] - mnew);
            const float p0 = __expf(s0[r] - mnew);
            const float p1 = __expf(s1[r] - mnew);
            float rs = p0 + p1;
            #pragma unroll
            for (int off = 8; off >= 1; off >>= 1)
                rs += __shfl_xor(rs, off, 32);
            lrow[r] = lrow[r] * scale + rs;
            mrow[r] = mnew;
            #pragma unroll
            for (int t = 0; t < 8; ++t) O[t][r] *= scale;
            const int prow = r + (lane >> 4) * 8;
            Pw[prow * 32 + laneIdx]      = (__bf16)p0;
            Pw[prow * 32 + 16 + laneIdx] = (__bf16)p1;
        }

        // ---- P (16x32) as A-fragment via per-wave LDS round-trip ----
        FragBF pf;
        {
            const __bf16* p = Pw + laneIdx * 32 + khalf;
            pf.u[0] = *(const v4u*)(p); pf.u[1] = *(const v4u*)(p + 16);
        }
        // ---- O += P @ V  (N = 128 in 8 tiles) ----
        const __bf16* Vc = &Vs[cur][0];
        #pragma unroll
        for (int nt = 0; nt < 8; ++nt) {
            FragBF vf;
            const __bf16* p = Vc + (size_t)(nt * 16 + laneIdx) * 32 + khalf;
            vf.u[0] = *(const v4u*)(p); vf.u[1] = *(const v4u*)(p + 16);
            O[nt] = __builtin_amdgcn_wmma_f32_16x16x32_bf16(
                false, pf.v, false, vf.v, (short)0, O[nt], false, false);
        }

        // TDM for next tile must be complete before anyone reads it next round
        if (wave == 0) __builtin_amdgcn_s_wait_tensorcnt(0);
        __syncthreads();
    }

    // ---- normalize + store bf16 to [B, L, D] with D-col = h*128 + d ----
    #pragma unroll
    for (int r = 0; r < 8; ++r) {
        const float inv = 1.0f / lrow[r];
        const int row = qBase + wave * 16 + r + (lane >> 4) * 8;
        const size_t base = ((size_t)(b * 2048 + row) * 4096) + h * 128;
        #pragma unroll
        for (int nt = 0; nt < 8; ++nt)
            Out[base + nt * 16 + laneIdx] = (__bf16)(O[nt][r] * inv);
    }
}

// ---- Host launch ------------------------------------------------------------
extern "C" void kernel_launch(void* const* d_in, const int* in_sizes, int n_in,
                              void* d_out, int out_size, void* d_ws, size_t ws_size,
                              hipStream_t stream)
{
    const float* x  = (const float*)d_in[0];
    const float* fc = (const float*)d_in[1];
    const float* fs = (const float*)d_in[2];
    const float* Wq = (const float*)d_in[3];
    const float* Wk = (const float*)d_in[4];
    const float* Wv = (const float*)d_in[5];
    const float* Wo = (const float*)d_in[6];
    float* out = (float*)d_out;

    const int M = 2 * 2048;   // B*L
    const int D = 4096;

    char* ws = (char*)d_ws;
    __bf16* qb = (__bf16*)(ws);                       // 32 MB  [B,H,L,HD]
    __bf16* kb = (__bf16*)(ws + ((size_t)32 << 20));  // 32 MB  [B,H,L,HD]
    __bf16* vb = (__bf16*)(ws + ((size_t)64 << 20));  // 32 MB  [B,H,L,HD]
    __bf16* at = (__bf16*)(ws + ((size_t)96 << 20));  // 32 MB  [B,L,D]

    dim3 grid(D / BLK_N, M / BLK_M);
    const float qscale = 0.08838834764831845f;  // 1/sqrt(128), folded into Q

    gemm_bt_kernel<1, 0><<<grid, 256, 0, stream>>>(x, Wq, nullptr, qb, fc, fs, M, D, D, qscale);
    gemm_bt_kernel<1, 0><<<grid, 256, 0, stream>>>(x, Wk, nullptr, kb, fc, fs, M, D, D, 1.0f);
    gemm_bt_kernel<2, 0><<<grid, 256, 0, stream>>>(x, Wv, nullptr, vb, nullptr, nullptr, M, D, D, 1.0f);

    attn_kernel<<<dim3(64, 16), 256, 0, stream>>>(qb, kb, vb, at);

    gemm_bt_kernel<0, 1><<<grid, 256, 0, stream>>>(at, Wo, out, nullptr, nullptr, nullptr, M, D, D, 1.0f);
}
